// PCTransformer_3951369912340
// MI455X (gfx1250) — matrix operand order repaired and checked by reading, not compiled, not run on hardware
//
#include <hip/hip_runtime.h>
#include <math.h>

// ---------------------------------------------------------------------------
// Types / helpers
// ---------------------------------------------------------------------------
typedef __bf16 bf16;
typedef __attribute__((ext_vector_type(16))) __bf16 bf16x16;
typedef __attribute__((ext_vector_type(8)))  __bf16 bf16x8;
typedef __attribute__((ext_vector_type(8)))  float  v8f;

#define BN 16
#define NN 1024
#define CC 384
#define HH 6
#define HD 64
#define KK 8
#define HID 768
#define ROWS (BN*NN)   // 16384

__device__ __forceinline__ bf16 f2b(float f) {
    union { float f; unsigned u; } x; x.f = f;
    unsigned r = x.u + 0x7fffu + ((x.u >> 16) & 1u);
    union { unsigned short s; bf16 b; } y; y.s = (unsigned short)(r >> 16);
    return y.b;
}
__device__ __forceinline__ float b2f(bf16 b) {
    union { unsigned short s; bf16 b; } y; y.b = b;
    union { unsigned u; float f; } x; x.u = ((unsigned)y.s) << 16;
    return x.f;
}
__device__ __forceinline__ bf16x16 ld_frag2(const bf16* p0, const bf16* p1) {
    union { bf16x16 v; bf16x8 h[2]; } u;
    u.h[0] = *(const bf16x8*)p0;
    u.h[1] = *(const bf16x8*)p1;
    return u.v;
}

// ---------------------------------------------------------------------------
// Weight packing: fp32 row-major [Kc,Nc] -> bf16 B-fragment tiles.
// Tile (kt,nt): lane l holds column nt*16+(l&15), K = kt*32+8*(l>>4)+(j<8 ? j : j+8).
// ---------------------------------------------------------------------------
__global__ void pack_weight_kernel(const float* __restrict__ W, bf16* __restrict__ out,
                                   int Kc, int Nc) {
    int tid = blockIdx.x * blockDim.x + threadIdx.x;
    int total = (Kc >> 5) * (Nc >> 4) * 32;
    if (tid >= total) return;
    int lane = tid & 31;
    int tile = tid >> 5;
    int tiles_n = Nc >> 4;
    int nt = tile % tiles_n, kt = tile / tiles_n;
    int col  = nt * 16 + (lane & 15);
    int krow = kt * 32 + 8 * (lane >> 4);
    bf16* dst = out + ((size_t)tile * 32 + lane) * 16;
    #pragma unroll
    for (int j = 0; j < 16; ++j)
        dst[j] = f2b(W[(size_t)(krow + (j < 8 ? j : j + 8)) * Nc + col]);
}

// ---------------------------------------------------------------------------
// LayerNorm (one wave per 384-wide row) -> bf16 out
// ---------------------------------------------------------------------------
__global__ void ln_kernel(const float* __restrict__ x, const float* __restrict__ w,
                          const float* __restrict__ bta, bf16* __restrict__ out) {
    int lane = threadIdx.x & 31;
    int row  = blockIdx.x * (blockDim.x >> 5) + (threadIdx.x >> 5);
    const float* xr = x + (size_t)row * CC;
    float v[12], s = 0.f, s2 = 0.f;
    #pragma unroll
    for (int i = 0; i < 12; ++i) { v[i] = xr[lane + i * 32]; s += v[i]; s2 += v[i] * v[i]; }
    #pragma unroll
    for (int m = 1; m < 32; m <<= 1) { s += __shfl_xor(s, m, 32); s2 += __shfl_xor(s2, m, 32); }
    float mean = s * (1.f / CC);
    float var  = s2 * (1.f / CC) - mean * mean;
    float rstd = rsqrtf(var + 1e-5f);
    bf16* orow = out + (size_t)row * CC;
    #pragma unroll
    for (int i = 0; i < 12; ++i) {
        int c = lane + i * 32;
        orow[c] = f2b((v[i] - mean) * rstd * w[c] + bta[c]);
    }
}

// ---------------------------------------------------------------------------
// Generic bf16 WMMA GEMM: C = act(A @ B + bias) [+ resid]
// One wave: 16 rows x 64 cols, double-buffered fragment loads.
// ACT: 0 = none, 1 = exact GELU.
// ---------------------------------------------------------------------------
template<int ACT>
__global__ void gemm_kernel(const bf16* __restrict__ A, const bf16* __restrict__ Bp,
                            const float* __restrict__ bias, const float* __restrict__ resid,
                            float* __restrict__ Cf, bf16* __restrict__ Cb, int ldcb,
                            int M, int Nc, int Kc) {
    int lane = threadIdx.x & 31;
    int w    = blockIdx.x * (blockDim.x >> 5) + (threadIdx.x >> 5);
    int tiles_n = Nc >> 4;
    int nquad   = tiles_n >> 2;
    int mt = w / nquad, nq = w % nquad;
    if (mt * 16 >= M) return;                    // wave-uniform
    int half = lane >> 4, ln = lane & 15;
    int kbA = half * 8;
    const bf16* arow = A + (size_t)(mt * 16 + ln) * Kc;
    const bf16* bbase = Bp + ((size_t)nq * 4 * 32 + lane) * 16;
    const size_t bstride = (size_t)tiles_n * 32 * 16;   // per k-chunk
    int KT = Kc >> 5;

    v8f acc[4] = {};
    bf16x16 a = ld_frag2(arow + kbA, arow + 16 + kbA);
    bf16x16 bf0 = *(const bf16x16*)(bbase + 0 * 512);
    bf16x16 bf1 = *(const bf16x16*)(bbase + 1 * 512);
    bf16x16 bf2 = *(const bf16x16*)(bbase + 2 * 512);
    bf16x16 bf3 = *(const bf16x16*)(bbase + 3 * 512);
    for (int kt = 0; kt < KT; ++kt) {
        int kn = (kt + 1 < KT) ? kt + 1 : kt;    // harmless reload on last iter
        int k0 = kn * 32;
        bf16x16 an = ld_frag2(arow + k0 + kbA, arow + k0 + 16 + kbA);
        const bf16* bn = bbase + (size_t)kn * bstride;
        bf16x16 c0 = *(const bf16x16*)(bn + 0 * 512);
        bf16x16 c1 = *(const bf16x16*)(bn + 1 * 512);
        bf16x16 c2 = *(const bf16x16*)(bn + 2 * 512);
        bf16x16 c3 = *(const bf16x16*)(bn + 3 * 512);
        acc[0] = __builtin_amdgcn_wmma_f32_16x16x32_bf16(false, a, false, bf0, (short)0, acc[0], false, false);
        acc[1] = __builtin_amdgcn_wmma_f32_16x16x32_bf16(false, a, false, bf1, (short)0, acc[1], false, false);
        acc[2] = __builtin_amdgcn_wmma_f32_16x16x32_bf16(false, a, false, bf2, (short)0, acc[2], false, false);
        acc[3] = __builtin_amdgcn_wmma_f32_16x16x32_bf16(false, a, false, bf3, (short)0, acc[3], false, false);
        a = an; bf0 = c0; bf1 = c1; bf2 = c2; bf3 = c3;
    }
    #pragma unroll
    for (int t = 0; t < 4; ++t) {
        int col = (nq * 4 + t) * 16 + ln;
        float bs = bias ? bias[col] : 0.f;
        #pragma unroll
        for (int r = 0; r < 8; ++r) {
            int row = mt * 16 + r + 8 * half;
            float vv = acc[t][r] + bs;
            if (ACT == 1) vv = 0.5f * vv * (1.f + erff(vv * 0.70710678118654752f));
            if (resid) vv += resid[(size_t)row * CC + col];
            if (Cf) Cf[(size_t)row * Nc + col] = vv;
            if (Cb) Cb[(size_t)row * ldcb + col] = f2b(vv);
        }
    }
}

// ---------------------------------------------------------------------------
// V transpose: qkv[b,n, 768 + h*64 + d] -> vt[((b*6+h)*64 + d)*1024 + n]
// ---------------------------------------------------------------------------
__global__ void transpose_v_kernel(const bf16* __restrict__ qkv, bf16* __restrict__ vt) {
    __shared__ bf16 tile[64 * 65];
    int t = threadIdx.x;               // 256
    int blk = blockIdx.x;              // B*H*16
    int b = blk / (HH * 16); int rem = blk % (HH * 16);
    int h = rem / 16; int n0 = (rem % 16) * 64;
    #pragma unroll
    for (int i = 0; i < 16; ++i) {
        int lin = t + i * 256;
        int n = lin >> 6, d = lin & 63;
        tile[d * 65 + n] = qkv[((size_t)(b * NN + n0 + n)) * 1152 + 768 + h * 64 + d];
    }
    __syncthreads();
    #pragma unroll
    for (int i = 0; i < 16; ++i) {
        int lin = t + i * 256;
        int d = lin >> 6, n = lin & 63;
        vt[(((size_t)(b * HH + h) * 64) + d) * NN + n0 + n] = tile[d * 65 + n];
    }
}

// ---------------------------------------------------------------------------
// Flash attention, transposed-S formulation. One wave per (b,h,16-query tile).
// S^T = K(16x32d) x Q^T(32d x 16q): C/D lane = query column, rows = keys ->
// softmax stats are lane-local; the P A-fragment for P@V is exactly the
// lane's own s0[j]/s1[j] registers. Two passes: (1) row max only (no
// cross-lane ops per chunk), (2) recompute scores, accumulate exp-sum + P@V.
// No LDS, no barriers, 2 shuffles + 8 broadcasts total per wave.
// ---------------------------------------------------------------------------
__global__ __launch_bounds__(128) void attn_kernel(const bf16* __restrict__ qkv,
                                                   const bf16* __restrict__ vt,
                                                   bf16* __restrict__ aout) {
    int lane = threadIdx.x & 31;
    int id = blockIdx.x * 4 + (threadIdx.x >> 5);
    int b = id / (HH * 64); int rem = id % (HH * 64);
    int h = rem >> 6; int mt = rem & 63;
    int half = lane >> 4, ln = lane & 15, kbA = half * 8;

    // Q^T B-fragments for d-chunks [0,32),[32,64); pre-scaled by HD^-0.5.
    bf16x16 qb[2];
    const bf16* qrow = qkv + ((size_t)(b * NN + mt * 16 + ln)) * 1152 + h * 64;
    #pragma unroll
    for (int c = 0; c < 2; ++c) {
        union { bf16x16 v; } u;
        u.v = ld_frag2(qrow + c * 32 + 8 * half, qrow + c * 32 + 16 + 8 * half);
        #pragma unroll
        for (int j = 0; j < 16; ++j) u.v[j] = f2b(b2f(u.v[j]) * 0.125f);
        qb[c] = u.v;
    }
    const bf16* kbase = qkv + (size_t)b * NN * 1152 + 384 + h * 64;
    const bf16* vbase = vt + ((size_t)(b * HH + h) * 64) * NN;

    // ---- pass 1: global row max (lane-local per chunk) ----
    float mloc = -1e30f;
    for (int jc = 0; jc < NN; jc += 32) {
        v8f s0 = {}, s1 = {};
        #pragma unroll
        for (int c = 0; c < 2; ++c) {
            const bf16* k0 = kbase + (size_t)(jc + ln) * 1152 + c * 32;
            bf16x16 a0 = ld_frag2(k0 + kbA, k0 + 16 + kbA);
            s0 = __builtin_amdgcn_wmma_f32_16x16x32_bf16(false, a0, false, qb[c], (short)0, s0, false, false);
            const bf16* k1 = kbase + (size_t)(jc + 16 + ln) * 1152 + c * 32;
            bf16x16 a1 = ld_frag2(k1 + kbA, k1 + 16 + kbA);
            s1 = __builtin_amdgcn_wmma_f32_16x16x32_bf16(false, a1, false, qb[c], (short)0, s1, false, false);
        }
        #pragma unroll
        for (int r = 0; r < 8; ++r)
            mloc = fmaxf(mloc, fmaxf(s0[r], s1[r]));
    }
    float m = fmaxf(mloc, __shfl_xor(mloc, 16, 32));

    // ---- pass 2: P = exp(S - m); denom + O = P @ V ----
    float sloc = 0.f;
    v8f o[4] = {};
    for (int jc = 0; jc < NN; jc += 32) {
        v8f s0 = {}, s1 = {};
        #pragma unroll
        for (int c = 0; c < 2; ++c) {
            const bf16* k0 = kbase + (size_t)(jc + ln) * 1152 + c * 32;
            bf16x16 a0 = ld_frag2(k0 + kbA, k0 + 16 + kbA);
            s0 = __builtin_amdgcn_wmma_f32_16x16x32_bf16(false, a0, false, qb[c], (short)0, s0, false, false);
            const bf16* k1 = kbase + (size_t)(jc + 16 + ln) * 1152 + c * 32;
            bf16x16 a1 = ld_frag2(k1 + kbA, k1 + 16 + kbA);
            s1 = __builtin_amdgcn_wmma_f32_16x16x32_bf16(false, a1, false, qb[c], (short)0, s1, false, false);
        }
        // lane (query q=l&15) holds keys jc+8*half+j (s0[j]) and
        // jc+16+8*half+j (s1[j]) == exactly the A-fragment element set.
        union { bf16x16 v; } pa;
        #pragma unroll
        for (int j = 0; j < 8; ++j) {
            float e0 = __expf(s0[j] - m);
            float e1 = __expf(s1[j] - m);
            sloc += e0 + e1;
            pa.v[j]     = f2b(e0);
            pa.v[8 + j] = f2b(e1);
        }
        #pragma unroll
        for (int t = 0; t < 4; ++t) {
            const bf16x16 vb = ld_frag2(vbase + (size_t)(t * 16 + ln) * NN + jc + 8 * half, vbase + (size_t)(t * 16 + ln) * NN + jc + 16 + 8 * half);
            o[t] = __builtin_amdgcn_wmma_f32_16x16x32_bf16(false, pa.v, false, vb, (short)0, o[t], false, false);
        }
    }
    float stot = sloc + __shfl_xor(sloc, 16, 32);
    float sden[8];
    #pragma unroll
    for (int r = 0; r < 8; ++r)
        sden[r] = __shfl(stot, r + 8 * half, 32);   // denom for query r+8*half
    #pragma unroll
    for (int t = 0; t < 4; ++t) {
        #pragma unroll
        for (int r = 0; r < 8; ++r) {
            int row = mt * 16 + r + 8 * half;
            aout[((size_t)(b * NN + row)) * CC + h * 64 + t * 16 + ln] = f2b(o[t][r] / sden[r]);
        }
    }
}

// ---------------------------------------------------------------------------
// KNN gather + leaky-ReLU + max over K -> cols [384,768) of cat.
// ---------------------------------------------------------------------------
__global__ void knn_kernel(const float* __restrict__ P1, const float* __restrict__ P2,
                           const int* __restrict__ knn_index, bf16* __restrict__ cat) {
    size_t idx = (size_t)blockIdx.x * blockDim.x + threadIdx.x;  // ROWS*CC
    int c = (int)(idx % CC);
    size_t row = idx / CC;
    int b = (int)(row / NN), n = (int)(row % NN);
    float center = P1[row * CC + c];
    float p2     = P2[row * CC + c];
    float best = -1e30f;
    #pragma unroll
    for (int k = 0; k < KK; ++k) {
        int g = knn_index[(size_t)(b * KK + k) * NN + n];
        float u = P1[(size_t)g * CC + c] - center + p2;
        u = (u > 0.f) ? u : 0.2f * u;
        best = fmaxf(best, u);
    }
    cat[row * HID + CC + c] = f2b(best);
}

// ---------------------------------------------------------------------------
// Host-side orchestration
// ---------------------------------------------------------------------------
static inline size_t align256(size_t x) { return (x + 255) & ~(size_t)255; }

extern "C" void kernel_launch(void* const* d_in, const int* in_sizes, int n_in,
                              void* d_out, int out_size, void* d_ws, size_t ws_size,
                              hipStream_t stream) {
    (void)in_sizes; (void)n_in; (void)out_size; (void)ws_size;
    const float* x      = (const float*)d_in[0];
    const int*   knn_ix = (const int*)  d_in[1];
    const float* ln1_w  = (const float*)d_in[2];
    const float* ln1_b  = (const float*)d_in[3];
    const float* w_qkv  = (const float*)d_in[4];
    const float* w_proj = (const float*)d_in[5];
    const float* b_proj = (const float*)d_in[6];
    const float* w_knn  = (const float*)d_in[7];
    const float* b_knn  = (const float*)d_in[8];
    const float* w_mrg  = (const float*)d_in[9];
    const float* b_mrg  = (const float*)d_in[10];
    const float* ln2_w  = (const float*)d_in[11];
    const float* ln2_b  = (const float*)d_in[12];
    const float* w_fc1  = (const float*)d_in[13];
    const float* b_fc1  = (const float*)d_in[14];
    const float* w_fc2  = (const float*)d_in[15];
    const float* b_fc2  = (const float*)d_in[16];
    float* out = (float*)d_out;

    char* p = (char*)d_ws;
    auto take = [&](size_t bytes) { char* r = p; p += align256(bytes); return r; };
    bf16*  nx_bf  = (bf16*) take((size_t)ROWS * CC * 2);
    bf16*  qkv_bf = (bf16*) take((size_t)ROWS * 1152 * 2);
    bf16*  vt_bf  = (bf16*) take((size_t)BN * HH * HD * NN * 2);
    bf16*  a_bf   = (bf16*) take((size_t)ROWS * CC * 2);
    bf16*  cat_bf = (bf16*) take((size_t)ROWS * HID * 2);
    float* P1     = (float*)take((size_t)ROWS * CC * 4);
    float* P2     = (float*)take((size_t)ROWS * CC * 4);
    float* x2     = (float*)take((size_t)ROWS * CC * 4);
    bf16*  h_bf   = (bf16*) take((size_t)ROWS * CC * 2);
    bf16*  fc1_bf = (bf16*) take((size_t)ROWS * HID * 2);
    bf16*  pk_qkv = (bf16*) take((size_t)CC * 1152 * 2);
    bf16*  pk_prj = (bf16*) take((size_t)CC * CC * 2);
    bf16*  pk_kn1 = (bf16*) take((size_t)CC * CC * 2);
    bf16*  pk_kn2 = (bf16*) take((size_t)CC * CC * 2);
    bf16*  pk_mrg = (bf16*) take((size_t)HID * CC * 2);
    bf16*  pk_fc1 = (bf16*) take((size_t)CC * HID * 2);
    bf16*  pk_fc2 = (bf16*) take((size_t)HID * CC * 2);

    auto pack = [&](const float* W, bf16* dst, int Kc, int Nc) {
        int total = (Kc / 32) * (Nc / 16) * 32;
        pack_weight_kernel<<<(total + 255) / 256, 256, 0, stream>>>(W, dst, Kc, Nc);
    };
    pack(w_qkv,           pk_qkv, CC, 1152);
    pack(w_proj,          pk_prj, CC, CC);
    pack(w_knn,           pk_kn1, CC, CC);           // rows 0..383  (feat-center)
    pack(w_knn + CC * CC, pk_kn2, CC, CC);           // rows 384..767 (center)
    pack(w_mrg,           pk_mrg, HID, CC);
    pack(w_fc1,           pk_fc1, CC, HID);
    pack(w_fc2,           pk_fc2, HID, CC);

    auto gemm = [&](int act, const bf16* A, const bf16* Bp, const float* bias,
                    const float* resid, float* Cf, bf16* Cb, int ldcb, int Nc, int Kc) {
        int waves = (ROWS / 16) * ((Nc / 16) / 4);
        int blocks = (waves + 7) / 8;
        if (act == 0)
            gemm_kernel<0><<<blocks, 256, 0, stream>>>(A, Bp, bias, resid, Cf, Cb, ldcb, ROWS, Nc, Kc);
        else
            gemm_kernel<1><<<blocks, 256, 0, stream>>>(A, Bp, bias, resid, Cf, Cb, ldcb, ROWS, Nc, Kc);
    };

    // 1) LN1 -> nx (bf16)
    ln_kernel<<<ROWS / 8, 256, 0, stream>>>(x, ln1_w, ln1_b, nx_bf);
    // 2) qkv = nx @ Wqkv
    gemm(0, nx_bf, pk_qkv, nullptr, nullptr, nullptr, qkv_bf, 1152, 1152, CC);
    // 3) V transpose
    transpose_v_kernel<<<BN * HH * 16, 256, 0, stream>>>(qkv_bf, vt_bf);
    // 4) attention -> a_bf
    attn_kernel<<<(BN * HH * 64) / 4, 128, 0, stream>>>(qkv_bf, vt_bf, a_bf);
    // 5) x1 = a_bf @ Wproj + b_proj -> cat[:, 0:384]
    gemm(0, a_bf, pk_prj, b_proj, nullptr, nullptr, cat_bf, HID, CC, CC);
    // 6) P1 = nx @ Wknn[0:C],  P2 = nx @ Wknn[C:2C] + b_knn
    gemm(0, nx_bf, pk_kn1, nullptr, nullptr, P1, nullptr, 0, CC, CC);
    gemm(0, nx_bf, pk_kn2, b_knn,  nullptr, P2, nullptr, 0, CC, CC);
    // 7) gather + leaky + max_k -> cat[:, 384:768]
    knn_kernel<<<((size_t)ROWS * CC) / 256, 256, 0, stream>>>(P1, P2, knn_ix, cat_bf);
    // 8) x2 = x + cat @ Wmerge + b_merge
    gemm(0, cat_bf, pk_mrg, b_mrg, x, x2, nullptr, 0, CC, HID);
    // 9) LN2 -> h (bf16)
    ln_kernel<<<ROWS / 8, 256, 0, stream>>>(x2, ln2_w, ln2_b, h_bf);
    // 10) fc1 + exact GELU
    gemm(1, h_bf, pk_fc1, b_fc1, nullptr, nullptr, fc1_bf, HID, HID, CC);
    // 11) out = x2 + fc1_out @ Wfc2 + b_fc2
    gemm(0, fc1_bf, pk_fc2, b_fc2, x2, out, nullptr, 0, CC, HID);
}
